// Pointnet2StructurePointNet_12506944766662
// MI455X (gfx1250) — matrix operand order repaired and checked by reading, compiled, not run on hardware
//
#include <hip/hip_runtime.h>
#include <hip/hip_bf16.h>

// ---------------------------------------------------------------------------
// PointNet++ MSG structure-point network for MI455X (gfx1250, wave32, WMMA).
//
//  * ~60 GFLOP total, all live data << 192MB L2 -> fuse grouping+MLP+maxpool
//    per sampled point entirely in LDS; never materialize [C, S*K] tensors.
//  * Weights pre-packed once per launch to f16 [cout][cinPad] so A-fragments
//    are two contiguous global_load_b128 per lane (no predication).
//  * Activations kept f16 in [m][C] layout so B-fragments are two contiguous
//    b128 loads and D-tiles store as one packed v8h per lane.
//  * All WMMA-executing control flow is wave-uniform (EXEC all-ones rule).
// ---------------------------------------------------------------------------

typedef __attribute__((ext_vector_type(16))) _Float16 v16h;
typedef __attribute__((ext_vector_type(8)))  _Float16 v8h;
typedef __attribute__((ext_vector_type(8)))  float    v8f;

#define BN_SCALE_C 0.99999500003749968f   // 1/sqrt(1+1e-5)

struct LayerP {
  const _Float16* wp;   // packed [cout][cinPad] f16, zero-padded
  const float* bi; const float* ga; const float* be;
  int cinPad; int cout;
};

__device__ __forceinline__ v16h cat8(v8h lo, v8h hi) {
  return __builtin_shufflevector(lo, hi, 0,1,2,3,4,5,6,7,8,9,10,11,12,13,14,15);
}

// A fragment (16x32 f16): lane -> row co0+(lane&15); K-half offset 8 for
// lanes 16-31.  Elements: K = koffA + {0..7} and koffA + 16 + {0..7}.
__device__ __forceinline__ v16h load_a_frag(const _Float16* __restrict__ wp,
                                            int cinPad, int co0, int ciBase, int lane)
{
  const int mrow  = lane & 15;
  const int koffA = (lane & 16) ? 8 : 0;
  const _Float16* p = wp + (size_t)(co0 + mrow) * cinPad + ciBase + koffA;
  return cat8(*(const v8h*)p, *(const v8h*)(p + 16));
}

// B fragment (32x16 f16) from [m][C] activations: lane -> col n0+(lane&15),
// K = koffB + {0..7} and koffB + 16 + {0..7}.
__device__ __forceinline__ v16h load_b_frag(const _Float16* __restrict__ act,
                                            int ldc, int n0, int ciBase, int lane)
{
  const int ncol  = n0 + (lane & 15);
  const int koffB = (lane & 16) ? 8 : 0;
  const _Float16* p = act + (size_t)ncol * ldc + ciBase + koffB;
  return cat8(*(const v8h*)p, *(const v8h*)(p + 16));
}

// One conv1x1+BN+ReLU layer, LDS f16 [k][cinPad] -> LDS f16 [k][cout].
__device__ __forceinline__ void run_layer_lds(const LayerP L,
    const _Float16* __restrict__ in, _Float16* __restrict__ outb,
    int K, int wave, int nwave, int lane)
{
  const int tilesN = K >> 4;
  const int tiles  = (L.cout >> 4) * tilesN;
  const int ncol16 = lane & 15;
  for (int t = wave; t < tiles; t += nwave) {          // wave-uniform loop
    const int tco = t / tilesN;
    const int tn  = t - tco * tilesN;
    const int co0 = tco << 4, n0 = tn << 4;
    v8f acc = {};
    for (int ciBase = 0; ciBase < L.cinPad; ciBase += 32) {
      const v16h a  = load_a_frag(L.wp, L.cinPad, co0, ciBase, lane);
      const v16h bb = load_b_frag(in, L.cinPad, n0, ciBase, lane);
      acc = __builtin_amdgcn_wmma_f32_16x16x32_f16(false, a, false, bb,
                                                   (short)0, acc, false, false);
    }
    const int cobase = co0 + ((lane & 16) ? 8 : 0);
    const int ncol   = n0 + ncol16;
    v8h hv;
#pragma unroll
    for (int r = 0; r < 8; ++r) {
      const int co = cobase + r;
      float v = (acc[r] + L.bi[co]) * (L.ga[co] * BN_SCALE_C) + L.be[co];
      hv[r] = (_Float16)fmaxf(v, 0.0f);
    }
    *(v8h*)(outb + (size_t)ncol * L.cout + cobase) = hv;
  }
}

// ---------------------------------------------------------------------------
// Weight repack: f32 [cout][cin] -> f16 [cout][cinPad] zero-padded.
// ---------------------------------------------------------------------------
__global__ void repack_w_kernel(const float* __restrict__ W,
    _Float16* __restrict__ Wp, int cin, int cinPad, int total)
{
  const int i = blockIdx.x * blockDim.x + threadIdx.x;
  if (i >= total) return;
  const int co = i / cinPad, ci = i - co * cinPad;
  Wp[i] = (ci < cin) ? (_Float16)W[(size_t)co * cin + ci] : (_Float16)0.0f;
}

// ---------------------------------------------------------------------------
// Farthest point sampling: one block per batch, LDS-resident running min-dist,
// tree argmax (tie -> lowest index, matching jnp.argmax).
// ---------------------------------------------------------------------------
__global__ void fps_kernel(const float* __restrict__ xyz, int* __restrict__ fidx,
                           int Npts, int S)
{
  __shared__ float dist[4096];
  __shared__ float rv[256];
  __shared__ int   ri[256];
  __shared__ int   curFar;
  const int b = blockIdx.x, tid = threadIdx.x, nthr = blockDim.x;
  const float* xb = xyz + (size_t)b * Npts * 3;
  for (int n = tid; n < Npts; n += nthr) dist[n] = 1e10f;
  if (tid == 0) curFar = 0;
  __syncthreads();
  for (int it = 0; it < S; ++it) {
    const int far = curFar;
    if (tid == 0) fidx[b * S + it] = far;
    const float cx = xb[far * 3 + 0], cy = xb[far * 3 + 1], cz = xb[far * 3 + 2];
    float bestV = -1.0f; int bestI = 0;
    for (int n = tid; n < Npts; n += nthr) {
      const float dx = xb[n * 3 + 0] - cx;
      const float dy = xb[n * 3 + 1] - cy;
      const float dz = xb[n * 3 + 2] - cz;
      float nd = fminf(dist[n], dx * dx + dy * dy + dz * dz);
      dist[n] = nd;
      if (nd > bestV) { bestV = nd; bestI = n; }
    }
    rv[tid] = bestV; ri[tid] = bestI;
    __syncthreads();
    for (int off = nthr >> 1; off > 0; off >>= 1) {
      if (tid < off) {
        if (rv[tid + off] > rv[tid] ||
            (rv[tid + off] == rv[tid] && ri[tid + off] < ri[tid])) {
          rv[tid] = rv[tid + off]; ri[tid] = ri[tid + off];
        }
      }
      __syncthreads();
    }
    if (tid == 0) curFar = ri[0];
    __syncthreads();
  }
}

__global__ void gather_xyz_kernel(const float* __restrict__ xyz,
    const int* __restrict__ fidx, float* __restrict__ nxyz,
    int Npts, int S, int total)
{
  const int i = blockIdx.x * blockDim.x + threadIdx.x;
  if (i >= total) return;
  const int b = i / S, s = i - b * S;
  const int id = fidx[b * S + s];
#pragma unroll
  for (int c = 0; c < 3; ++c)
    nxyz[(size_t)(b * S + s) * 3 + c] = xyz[((size_t)b * Npts + id) * 3 + c];
}

// Ball query: first K in-ball points in index order, pad with first hit.
__global__ void ball_query_kernel(const float* __restrict__ xyz,
    const float* __restrict__ nxyz, int* __restrict__ gidx,
    int Npts, int S, int K, float r2, int Bn)
{
  const int i = blockIdx.x * blockDim.x + threadIdx.x;
  if (i >= Bn * S) return;
  const int b = i / S, s = i - b * S;
  const float cx = nxyz[(size_t)(b * S + s) * 3 + 0];
  const float cy = nxyz[(size_t)(b * S + s) * 3 + 1];
  const float cz = nxyz[(size_t)(b * S + s) * 3 + 2];
  int* gout = gidx + ((size_t)b * S + s) * K;
  const float* xb = xyz + (size_t)b * Npts * 3;
  int cnt = 0, first = -1;
  for (int n = 0; n < Npts; ++n) {
    const float dx = xb[n * 3 + 0] - cx;
    const float dy = xb[n * 3 + 1] - cy;
    const float dz = xb[n * 3 + 2] - cz;
    if (dx * dx + dy * dy + dz * dz <= r2) {
      if (first < 0) first = n;
      if (cnt < K) gout[cnt] = n;
      ++cnt;
    }
  }
  if (first < 0) first = 0;
  for (int k = (cnt < K ? cnt : K); k < K; ++k) gout[k] = first;
}

// ---------------------------------------------------------------------------
// Fused MSG branch: one workgroup per (sample s, batch b).
// Gather K neighbors into LDS f16 [k][cinPad], run 3 conv+BN+ReLU layers as
// LDS-resident WMMA GEMMs ping-ponging two buffers, maxpool over K, write
// f16 [B][S][Ctot] features.
// ---------------------------------------------------------------------------
__global__ void msg_branch_kernel(
    const float*    __restrict__ xyz,    // [B, Npts, 3] f32
    const _Float16* __restrict__ feats,  // [B, Npts, Cf] f16 or nullptr
    int Cf,
    const float* __restrict__ nxyz,      // [B, S, 3] f32
    const int*   __restrict__ gidx,      // [B, S, K]
    int Npts, int S, int K,
    LayerP L1, LayerP L2, LayerP L3,
    int maxC,
    _Float16* __restrict__ out,          // [B, S, Ctot] f16
    int Ctot, int coff)
{
  extern __shared__ char smem[];
  _Float16* buf0 = (_Float16*)smem;                 // [K * maxC]
  _Float16* buf1 = buf0 + (size_t)maxC * K;         // [K * maxC]
  int*      gsh  = (int*)(buf1 + (size_t)maxC * K); // [K]

  const int b = blockIdx.y, s = blockIdx.x;
  const int tid = threadIdx.x, nthr = blockDim.x;
  const int lane = tid & 31, wave = tid >> 5, nwave = nthr >> 5;

  for (int k = tid; k < K; k += nthr)
    gsh[k] = gidx[((size_t)b * S + s) * K + k];
  __syncthreads();

  const float cx = nxyz[(size_t)(b * S + s) * 3 + 0];
  const float cy = nxyz[(size_t)(b * S + s) * 3 + 1];
  const float cz = nxyz[(size_t)(b * S + s) * 3 + 2];

  // Gather into buf0 as [k][ciPad0] (channels >= cin are zero padding).
  const int ciPad0 = L1.cinPad;
  const int total = K * ciPad0;
  for (int e = tid; e < total; e += nthr) {
    const int k = e / ciPad0, c = e - k * ciPad0;
    const int id = gsh[k];
    _Float16 h = (_Float16)0.0f;
    if (feats != nullptr) {
      if (c < Cf) {
        h = feats[((size_t)b * Npts + id) * Cf + c];
      } else if (c < Cf + 3) {
        const int d = c - Cf;
        const float ctr = (d == 0) ? cx : (d == 1) ? cy : cz;
        h = (_Float16)(xyz[((size_t)b * Npts + id) * 3 + d] - ctr);
      }
    } else if (c < 3) {
      const float ctr = (c == 0) ? cx : (c == 1) ? cy : cz;
      h = (_Float16)(xyz[((size_t)b * Npts + id) * 3 + c] - ctr);
    }
    buf0[e] = h;
  }
  __syncthreads();

  run_layer_lds(L1, buf0, buf1, K, wave, nwave, lane);
  __syncthreads();
  run_layer_lds(L2, buf1, buf0, K, wave, nwave, lane);
  __syncthreads();
  run_layer_lds(L3, buf0, buf1, K, wave, nwave, lane);
  __syncthreads();

  // Maxpool over K neighbors; ReLU output so max >= 0.
  for (int c = tid; c < L3.cout; c += nthr) {
    float m = 0.0f;
    for (int k = 0; k < K; ++k)
      m = fmaxf(m, (float)buf1[(size_t)k * L3.cout + c]);
    out[((size_t)b * S + s) * Ctot + coff + c] = (_Float16)m;
  }
}

// ---------------------------------------------------------------------------
// Generic conv1x1 + BN (+f16 residual) (+ReLU) GEMM.
//  X : f16 [B][M][cinPad];  Yh : f16 [B][M][cout]  OR  Yf : f32 [B][cout][M].
// One 16x16 WMMA tile per wave, wave-uniform control flow.
// ---------------------------------------------------------------------------
__global__ void conv_bn_gemm_kernel(const _Float16* __restrict__ X,
    LayerP L,
    const _Float16* __restrict__ Rres,   // [B][M][cout] or nullptr
    _Float16* __restrict__ Yh, float* __restrict__ Yf,
    int M, int Bn, int relu)
{
  const int lane = threadIdx.x & 31;
  const int wave = threadIdx.x >> 5;
  const int tilesN = M >> 4;
  const int tilesPerB = (L.cout >> 4) * tilesN;
  const int gt = blockIdx.x * (blockDim.x >> 5) + wave;   // wave-uniform
  if (gt >= Bn * tilesPerB) return;
  const int b = gt / tilesPerB;
  const int t = gt - b * tilesPerB;
  const int tco = t / tilesN;
  const int tn  = t - tco * tilesN;
  const int co0 = tco << 4, n0 = tn << 4;
  const _Float16* Xb = X + (size_t)b * M * L.cinPad;
  v8f acc = {};
  for (int ciBase = 0; ciBase < L.cinPad; ciBase += 32) {
    const v16h a  = load_a_frag(L.wp, L.cinPad, co0, ciBase, lane);
    const v16h bb = load_b_frag(Xb, L.cinPad, n0, ciBase, lane);
    acc = __builtin_amdgcn_wmma_f32_16x16x32_f16(false, a, false, bb,
                                                 (short)0, acc, false, false);
  }
  const int cobase = co0 + ((lane & 16) ? 8 : 0);
  const int ncol   = n0 + (lane & 15);
  if (Yh) {
    v8h hv;
#pragma unroll
    for (int r = 0; r < 8; ++r) {
      const int co = cobase + r;
      float v = (acc[r] + L.bi[co]) * (L.ga[co] * BN_SCALE_C) + L.be[co];
      if (Rres) v += (float)Rres[((size_t)b * M + ncol) * L.cout + co];
      if (relu) v = fmaxf(v, 0.0f);
      hv[r] = (_Float16)v;
    }
    *(v8h*)(Yh + ((size_t)b * M + ncol) * L.cout + cobase) = hv;
  } else {
#pragma unroll
    for (int r = 0; r < 8; ++r) {
      const int co = cobase + r;
      float v = (acc[r] + L.bi[co]) * (L.ga[co] * BN_SCALE_C) + L.be[co];
      if (Rres) v += (float)Rres[((size_t)b * M + ncol) * L.cout + co];
      if (relu) v = fmaxf(v, 0.0f);
      Yf[((size_t)b * L.cout + co) * M + ncol] = v;
    }
  }
}

// Row softmax over Ncols, one wave32 per row.
__global__ void softmax_rows_kernel(float* __restrict__ pm, int Rows, int Ncols)
{
  const int row = blockIdx.x * (blockDim.x >> 5) + (threadIdx.x >> 5);
  if (row >= Rows) return;
  const int lane = threadIdx.x & 31;
  float* p = pm + (size_t)row * Ncols;
  float mx = -1e30f;
  for (int j = lane; j < Ncols; j += 32) mx = fmaxf(mx, p[j]);
  for (int o = 16; o > 0; o >>= 1) mx = fmaxf(mx, __shfl_xor(mx, o, 32));
  float sum = 0.0f;
  for (int j = lane; j < Ncols; j += 32) { float e = __expf(p[j] - mx); p[j] = e; sum += e; }
  for (int o = 16; o > 0; o >>= 1) sum += __shfl_xor(sum, o, 32);
  const float inv = 1.0f / sum;
  for (int j = lane; j < Ncols; j += 32) p[j] *= inv;
}

// structure_points[b][s][c] = sum_n pm[b][s][n] * xyzn[b][n][c]
__global__ void structpts_kernel(const float* __restrict__ pm,
    const float* __restrict__ xyzn, float* __restrict__ sp,
    int Sp, int Nn, int total)
{
  const int i = blockIdx.x * blockDim.x + threadIdx.x;
  if (i >= total) return;
  const int c = i % 3;
  const int rest = i / 3;
  const int s = rest % Sp;
  const int b = rest / Sp;
  const float* prow = pm + ((size_t)b * Sp + s) * Nn;
  const float* xcol = xyzn + (size_t)b * Nn * 3 + c;
  float acc = 0.0f;
  for (int n = 0; n < Nn; ++n) acc += prow[n] * xcol[(size_t)n * 3];
  sp[i] = acc;
}

// ---------------------------------------------------------------------------
// Host-side orchestration.
// ---------------------------------------------------------------------------
struct RawCB { const float* w; const float* b; const float* g; const float* be; int cin; int cout; };

extern "C" void kernel_launch(void* const* d_in, const int* in_sizes, int n_in,
                              void* d_out, int out_size, void* d_ws, size_t ws_size,
                              hipStream_t stream)
{
  (void)in_sizes; (void)n_in; (void)out_size; (void)ws_size;
  const int B = 8, N = 4096;

  // ---- input unpack (pointcloud, then conv_bn params in w,b,g,be order) ----
  int p = 0;
  const float* pc = (const float*)d_in[p++];
  auto take = [&](int cin, int cout) -> RawCB {
    RawCB r;
    r.w  = (const float*)d_in[p++];
    r.b  = (const float*)d_in[p++];
    r.g  = (const float*)d_in[p++];
    r.be = (const float*)d_in[p++];
    r.cin = cin; r.cout = cout;
    return r;
  };
  RawCB sa1r[3][3];
  { int dims[3][4] = {{3,32,32,64},{3,64,64,128},{3,64,96,128}};
    for (int br = 0; br < 3; ++br)
      for (int l = 0; l < 3; ++l) sa1r[br][l] = take(dims[br][l], dims[br][l+1]); }
  RawCB sa2r[3][3];
  { int dims[3][4] = {{323,64,64,128},{323,128,128,256},{323,128,128,256}};
    for (int br = 0; br < 3; ++br)
      for (int l = 0; l < 3; ++l) sa2r[br][l] = take(dims[br][l], dims[br][l+1]); }
  RawCB rfr[5];
  { int dims[6] = {640,512,256,128,64,64};
    for (int l = 0; l < 5; ++l) rfr[l] = take(dims[l], dims[l+1]); }
  RawCB mc1r = take(64, 128);
  RawCB mc2r = take(128, 64);

  // ---- workspace carve ----
  char* wp = (char*)d_ws;
  auto alloc = [&](size_t bytes) -> void* {
    void* r = (void*)wp; wp += (bytes + 255) & ~(size_t)255; return r;
  };

  // packed f16 weights + repack launches
  auto makeL = [&](RawCB r) -> LayerP {
    const int cp = (r.cin + 31) & ~31;
    _Float16* w16 = (_Float16*)alloc((size_t)r.cout * cp * sizeof(_Float16));
    const int tot = r.cout * cp;
    repack_w_kernel<<<(tot + 255) / 256, 256, 0, stream>>>(r.w, w16, r.cin, cp, tot);
    LayerP L; L.wp = w16; L.bi = r.b; L.ga = r.g; L.be = r.be;
    L.cinPad = cp; L.cout = r.cout;
    return L;
  };
  LayerP sa1[3][3], sa2[3][3], rf[5], mc1, mc2;
  for (int br = 0; br < 3; ++br) for (int l = 0; l < 3; ++l) sa1[br][l] = makeL(sa1r[br][l]);
  for (int br = 0; br < 3; ++br) for (int l = 0; l < 3; ++l) sa2[br][l] = makeL(sa2r[br][l]);
  for (int l = 0; l < 5; ++l) rf[l] = makeL(rfr[l]);
  mc1 = makeL(mc1r);
  mc2 = makeL(mc2r);

  int*   fidx1 = (int*)alloc((size_t)B * 512 * 4);
  float* nx1   = (float*)alloc((size_t)B * 512 * 3 * 4);
  int K1[3] = {16, 32, 128};  float R1[3] = {0.1f, 0.2f, 0.4f};
  int* g1[3];
  for (int i = 0; i < 3; ++i) g1[i] = (int*)alloc((size_t)B * 512 * K1[i] * 4);
  _Float16* feats1 = (_Float16*)alloc((size_t)B * 512 * 320 * sizeof(_Float16)); // [B][512][320]
  int*   fidx2 = (int*)alloc((size_t)B * 128 * 4);
  float* nx2   = (float*)alloc((size_t)B * 128 * 3 * 4);
  int K2[3] = {32, 64, 128};  float R2[3] = {0.2f, 0.4f, 0.8f};
  int* g2[3];
  for (int i = 0; i < 3; ++i) g2[i] = (int*)alloc((size_t)B * 128 * K2[i] * 4);
  _Float16* feats2 = (_Float16*)alloc((size_t)B * 128 * 640 * sizeof(_Float16)); // [B][128][640]
  _Float16* rb0h = (_Float16*)alloc((size_t)B * 128 * 512 * sizeof(_Float16));
  _Float16* rb1h = (_Float16*)alloc((size_t)B * 128 * 512 * sizeof(_Float16));
  _Float16* y1h  = (_Float16*)alloc((size_t)B * 128 * 128 * sizeof(_Float16));

  float* outF    = (float*)d_out;
  float* sp_out  = outF;                 // [8,64,3]
  float* xyz_out = outF + 1536;          // [8,128,3]
  float* pm_out  = outF + 1536 + 3072;   // [8,64,128]

  // ---- SA1 ----
  fps_kernel<<<dim3(B), dim3(256), 0, stream>>>(pc, fidx1, N, 512);
  { int tot = B * 512;
    gather_xyz_kernel<<<(tot + 255) / 256, 256, 0, stream>>>(pc, fidx1, nx1, N, 512, tot); }
  for (int i = 0; i < 3; ++i) {
    int tot = B * 512;
    ball_query_kernel<<<(tot + 255) / 256, 256, 0, stream>>>(
        pc, nx1, g1[i], N, 512, K1[i], R1[i] * R1[i], B);
  }
  { int coff = 0;
    for (int i = 0; i < 3; ++i) {
      LayerP* Ls = sa1[i];
      int maxC = Ls[0].cinPad;
      for (int l = 0; l < 3; ++l) if (Ls[l].cout > maxC) maxC = Ls[l].cout;
      size_t sh = (size_t)2 * maxC * K1[i] * sizeof(_Float16) + (size_t)K1[i] * sizeof(int);
      msg_branch_kernel<<<dim3(512, B), dim3(256), sh, stream>>>(
          pc, nullptr, 0, nx1, g1[i], N, 512, K1[i],
          Ls[0], Ls[1], Ls[2], maxC, feats1, 320, coff);
      coff += Ls[2].cout;
    } }

  // ---- SA2 ----
  fps_kernel<<<dim3(B), dim3(256), 0, stream>>>(nx1, fidx2, 512, 128);
  { int tot = B * 128;
    gather_xyz_kernel<<<(tot + 255) / 256, 256, 0, stream>>>(nx1, fidx2, nx2, 512, 128, tot); }
  for (int i = 0; i < 3; ++i) {
    int tot = B * 128;
    ball_query_kernel<<<(tot + 255) / 256, 256, 0, stream>>>(
        nx1, nx2, g2[i], 512, 128, K2[i], R2[i] * R2[i], B);
  }
  { int coff = 0;
    for (int i = 0; i < 3; ++i) {
      LayerP* Ls = sa2[i];
      int maxC = Ls[0].cinPad;                     // 352
      for (int l = 0; l < 3; ++l) if (Ls[l].cout > maxC) maxC = Ls[l].cout;
      size_t sh = (size_t)2 * maxC * K2[i] * sizeof(_Float16) + (size_t)K2[i] * sizeof(int);
      msg_branch_kernel<<<dim3(128, B), dim3(256), sh, stream>>>(
          nx1, feats1, 320, nx2, g2[i], 512, 128, K2[i],
          Ls[0], Ls[1], Ls[2], maxC, feats2, 640, coff);
      coff += Ls[2].cout;
    } }

  // ---- refine + merge (generic WMMA GEMM, f16 chained activations) ----
  auto gemm = [&](const _Float16* X, const _Float16* res, _Float16* Yh, float* Yf,
                  LayerP L, int M, int relu) {
    const int tiles = B * (L.cout / 16) * (M / 16);
    const int wavesPerBlk = 4;
    const int blocks = (tiles + wavesPerBlk - 1) / wavesPerBlk;
    conv_bn_gemm_kernel<<<blocks, wavesPerBlk * 32, 0, stream>>>(X, L, res, Yh, Yf, M, B, relu);
  };
  gemm(feats2, nullptr, rb0h, nullptr, rf[0], 128, 1);
  gemm(rb0h,   nullptr, rb1h, nullptr, rf[1], 128, 1);
  gemm(rb1h,   nullptr, rb0h, nullptr, rf[2], 128, 1);
  gemm(rb0h,   nullptr, rb1h, nullptr, rf[3], 128, 1);
  gemm(rb1h,   nullptr, rb0h, nullptr, rf[4], 128, 1);   // h: f16 [B][128][64] in rb0h
  gemm(rb0h,   nullptr, y1h,  nullptr, mc1,  128, 1);    // relu(bn(conv 64->128))
  gemm(y1h,    rb0h,    nullptr, pm_out, mc2, 128, 1);   // relu(bn(conv 128->64)+h), f32 [B][64][128]

  // ---- softmax over N=128, per (b, s) row ----
  { const int rows = B * 64;
    softmax_rows_kernel<<<(rows + 7) / 8, 256, 0, stream>>>(pm_out, rows, 128); }

  // ---- outputs: xyz_n copy + structure points ----
  hipMemcpyAsync(xyz_out, nx2, (size_t)B * 128 * 3 * 4, hipMemcpyDeviceToDevice, stream);
  { const int tot = B * 64 * 3;
    structpts_kernel<<<(tot + 255) / 256, 256, 0, stream>>>(pm_out, nx2, sp_out, 64, 128, tot); }
}
